// Attention_51470888075870
// MI455X (gfx1250) — compile-verified, matrix operand order checked by reading, not run on hardware
//
#include <hip/hip_runtime.h>

#define EPS 1e-5f

typedef _Float16 h16;
typedef __attribute__((ext_vector_type(16))) _Float16 v16h;
typedef __attribute__((ext_vector_type(8)))  float    v8f;

// problem dims
#define Bq 16
#define Cc 64
#define Nn 200
#define Tt 64
#define HC 32
#define C4 256
#define KK 768      // 4c * 3 im2col depth

// LDS / workspace strides (all multiples of 8 halves -> 16B row alignment)
#define T_LD  776   // im2colT row stride (halves)
#define W_LD  776
#define QT_LD 40    // q/k/attn_out transposed [t][hc]
#define AT_LD 66    // f32 attn / out accumulator stride
#define AH_LD 72

// f16 weight blob offsets in workspace (halves)
#define OFF_QW 0
#define OFF_KW 24832
#define OFF_VW 49664      // vW expanded to [32][776] (k%3==2 carries value)
#define OFF_PK 74496      // packed oW(64x40) + ffW1(64x72)@2560 + ffW2@7168
#define PK_HALVES 11776

static __device__ inline v8f wmma16(v16h a, v16h b, v8f c) {
  return __builtin_amdgcn_wmma_f32_16x16x32_f16(false, a, false, b, (short)0, c,
                                                false, false);
}

// Fragment loader (ISA 7.12.2): lane&15 = leading index, K contiguous per lane.
// lanes 0-15 hold K {0..7,16..23}+col0, lanes 16-31 hold K {8..15,24..31}+col0.
// Used for both A (lane = M) and B-from-[N][K] storage (lane = N).
static __device__ inline v16h fragA(const h16* base, int ld, int row0, int col0,
                                    int lane) {
  int m = (lane & 15) + row0;
  int ko = (lane & 16) ? 8 : 0;
  v16h f;
#pragma unroll
  for (int v = 0; v < 8; ++v) {
    int kb = ((v < 4) ? 2 * v : 16 + 2 * (v - 4)) + ko + col0;
    const h16* p = base + m * ld + kb;
    f[2 * v] = p[0];
    f[2 * v + 1] = p[1];
  }
  return f;
}

// C/D layout: VGPR r -> M = row0 + r + 8*(lane>=16), N = col0 + (lane&15)
static __device__ inline void storeD_hT(h16* base, int ld, int row0, int col0,
                                        int lane, v8f d) {  // writes [N][M]
  int nn = (lane & 15) + col0;
  int mo = (lane & 16) ? 8 : 0;
#pragma unroll
  for (int r = 0; r < 8; ++r) base[nn * ld + row0 + mo + r] = (h16)d[r];
}
static __device__ inline void storeD_h(h16* base, int ld, int row0, int col0,
                                       int lane, v8f d) {   // writes [M][N]
  int nn = (lane & 15) + col0;
  int mo = (lane & 16) ? 8 : 0;
#pragma unroll
  for (int r = 0; r < 8; ++r) base[(row0 + mo + r) * ld + nn] = (h16)d[r];
}
static __device__ inline void storeD_f(float* base, int ld, int row0, int col0,
                                       int lane, v8f d) {
  int nn = (lane & 15) + col0;
  int mo = (lane & 16) ? 8 : 0;
#pragma unroll
  for (int r = 0; r < 8; ++r) base[(row0 + mo + r) * ld + nn] = d[r];
}
static __device__ inline v8f loadC_f(const float* base, int ld, int row0,
                                     int col0, int lane) {
  int nn = (lane & 15) + col0;
  int mo = (lane & 16) ? 8 : 0;
  v8f c;
#pragma unroll
  for (int r = 0; r < 8; ++r) c[r] = base[(row0 + mo + r) * ld + nn];
  return c;
}

// CDNA5 async global->LDS copy (16B) + join (ISA §10, §15.18.3).
// Generic LDS pointer low 32 bits == LDS address (aperture rule §10.2).
static __device__ inline void async_ld16(h16* ldsp, const h16* gp) {
  uint32_t loff = (uint32_t)(uintptr_t)ldsp;
  asm volatile("global_load_async_to_lds_b128 %0, %1, off" ::"v"(loff), "v"(gp)
               : "memory");
}
static __device__ inline void wait_async0() {
  asm volatile("s_wait_asynccnt 0x0" ::: "memory");
}

// ---------------- weight convert kernel (once) ----------------
__global__ __launch_bounds__(256) void cvt_w(
    const float* __restrict__ qW, const float* __restrict__ kW,
    const float* __restrict__ vW, const float* __restrict__ oW,
    const float* __restrict__ ffW1, const float* __restrict__ ffW2,
    h16* __restrict__ wh) {
  int i = blockIdx.x * 256 + threadIdx.x;  // grid covers 24832
  if (i < 24832) {
    int r = i / W_LD, c = i % W_LD;
    float q = (c < KK) ? qW[r * KK + c] : 0.f;
    float k = (c < KK) ? kW[r * KK + c] : 0.f;
    float v = 0.f;
    if (c < KK && (c % 3) == 2) v = vW[r * C4 + c / 3];
    wh[OFF_QW + i] = (h16)q;
    wh[OFF_KW + i] = (h16)k;
    wh[OFF_VW + i] = (h16)v;
  }
  if (i < PK_HALVES) {
    h16 val;
    if (i < 2560) {
      int r = i / 40, c = i % 40;
      val = (h16)((c < HC) ? oW[r * HC + c] : 0.f);
    } else if (i < 7168) {
      int j = i - 2560, r = j / 72, c = j % 72;
      val = (h16)((c < Cc) ? ffW1[r * Cc + c] : 0.f);
    } else {
      int j = i - 7168, r = j / 72, c = j % 72;
      val = (h16)((c < Cc) ? ffW2[r * Cc + c] : 0.f);
    }
    wh[OFF_PK + i] = val;
  }
}

// ---------------- stats kernels ----------------
__global__ __launch_bounds__(256) void stats_tn(const float* __restrict__ x,
                                                float* __restrict__ mu,
                                                float* __restrict__ rs) {
  int ch = blockIdx.x;
  int ci = ch / Nn, ni = ch % Nn;
  int tid = threadIdx.x;
  float s = 0.f, q = 0.f;
  for (int e = tid; e < Bq * Tt; e += 256) {
    int bi = e >> 6, ti = e & 63;
    float v = x[(((size_t)bi * Cc + ci) * Nn + ni) * Tt + ti];
    s += v;
    q += v * v;
  }
  __shared__ float rsum[256], rsq[256];
  rsum[tid] = s;
  rsq[tid] = q;
  __syncthreads();
  for (int off = 128; off > 0; off >>= 1) {
    if (tid < off) {
      rsum[tid] += rsum[tid + off];
      rsq[tid] += rsq[tid + off];
    }
    __syncthreads();
  }
  if (tid == 0) {
    float m = rsum[0] * (1.f / (Bq * Tt));
    float v = rsq[0] * (1.f / (Bq * Tt)) - m * m;
    mu[ch] = m;
    rs[ch] = rsqrtf(v + EPS);
  }
}

__global__ __launch_bounds__(256) void stats_sn(const float* __restrict__ x,
                                                float* __restrict__ mu,
                                                float* __restrict__ rs) {
  int gid = blockIdx.x * 256 + threadIdx.x;  // bi*4096 + ci*64 + ti
  if (gid >= Bq * Cc * Tt) return;
  int bi = gid >> 12;
  int rem = gid & 4095;
  int ci = rem >> 6, ti = rem & 63;
  const float* base = x + (((size_t)bi * Cc + ci) * Nn) * Tt + ti;
  float s = 0.f, q = 0.f;
  for (int nidx = 0; nidx < Nn; ++nidx) {
    float v = base[nidx * Tt];
    s += v;
    q += v * v;
  }
  float m = s * (1.f / Nn);
  float var = q * (1.f / Nn) - m * m;
  mu[gid] = m;
  rs[gid] = rsqrtf(var + EPS);
}

__global__ __launch_bounds__(256) void stats_stn(const float* __restrict__ x,
                                                 float* __restrict__ mu,
                                                 float* __restrict__ rs) {
  int bc = blockIdx.x;
  const float* base = x + (size_t)bc * Nn * Tt;
  int tid = threadIdx.x;
  float s = 0.f, q = 0.f;
  for (int e = tid; e < Nn * Tt; e += 256) {
    float v = base[e];
    s += v;
    q += v * v;
  }
  __shared__ float rsum[256], rsq[256];
  rsum[tid] = s;
  rsq[tid] = q;
  __syncthreads();
  for (int off = 128; off > 0; off >>= 1) {
    if (tid < off) {
      rsum[tid] += rsum[tid + off];
      rsq[tid] += rsq[tid + off];
    }
    __syncthreads();
  }
  if (tid == 0) {
    float m = rsum[0] * (1.f / (Nn * Tt));
    float v = rsq[0] * (1.f / (Nn * Tt)) - m * m;
    mu[bc] = m;
    rs[bc] = rsqrtf(v + EPS);
  }
}

// ---------------- fused per-sequence kernel ----------------
__global__ __launch_bounds__(256) void attn_seq(
    const float* __restrict__ x, const float* __restrict__ tn_g,
    const float* __restrict__ tn_b, const float* __restrict__ sn_g,
    const float* __restrict__ sn_b, const float* __restrict__ stn_g,
    const float* __restrict__ stn_b, const float* __restrict__ ffb1,
    const float* __restrict__ ffb2, const float* __restrict__ tn_mu,
    const float* __restrict__ tn_rs, const float* __restrict__ sn_mu,
    const float* __restrict__ sn_rs, const float* __restrict__ stn_mu,
    const float* __restrict__ stn_rs, const h16* __restrict__ wh,
    float* __restrict__ out) {
  __shared__ __align__(16) h16 sT[Tt * T_LD];     // 99.3 KB im2col^T [t][k]
  __shared__ __align__(16) h16 sWa[32 * W_LD];    // 49.7 KB async weight buf A
  __shared__ __align__(16) h16 sWb[32 * W_LD];    // 49.7 KB async weight buf B
  __shared__ __align__(16) h16 sQt[Tt * QT_LD];   // q^T then attn_out^T
  __shared__ __align__(16) h16 sKt[Tt * QT_LD];   // k^T
  __shared__ __align__(16) h16 sV[HC * AH_LD];    // v row-major [h][t]
  __shared__ __align__(16) float sAttn[Tt * AT_LD];
  __shared__ __align__(16) h16 sAH[Tt * AH_LD];   // probs [i][j]; then out_f^T
  __shared__ __align__(16) float sOut[Cc * AT_LD];
  __shared__ __align__(16) h16 sHt[Tt * AH_LD];   // relu hidden ^T [t][d]

  const int tid = threadIdx.x;
  const int lane = tid & 31, wave = tid >> 5;
  const int seq = blockIdx.x;
  const int bi = seq / Nn, ni = seq % Nn;

  // kick off async DMA of qW -> sWa, kW -> sWb (overlaps normalize pass)
  for (int e = tid; e < 3104; e += 256) {  // 24832 halves = 3104 x 16B
    async_ld16(&sWa[e * 8], wh + OFF_QW + e * 8);
    async_ld16(&sWb[e * 8], wh + OFF_KW + e * 8);
  }

  // causal zero-pad: rows 0,1 of im2colT
  for (int e = tid; e < 2 * T_LD; e += 256) sT[e] = (h16)0.f;
  __syncthreads();

  // load input, apply 3 norms, scatter into im2colT:
  // value x_f[c4][s] -> sT[s+2-dk][c4*3+dk] for dk=0..2
  for (int e = tid; e < Cc * Tt; e += 256) {
    int ci = e >> 6, ti = e & 63;
    float raw = x[(((size_t)bi * Cc + ci) * Nn + ni) * Tt + ti];
    int ch = ci * Nn + ni;
    float xt = (raw - tn_mu[ch]) * tn_rs[ch] * tn_g[ch] + tn_b[ch];
    int si = (bi * Cc + ci) * Tt + ti;
    float xs = (raw - sn_mu[si]) * sn_rs[si] * sn_g[ci] + sn_b[ci];
    int sti = bi * Cc + ci;
    float xst = (raw - stn_mu[sti]) * stn_rs[sti] * stn_g[ci] + stn_b[ci];
    float vals[4] = {raw, xt, xs, xst};
#pragma unroll
    for (int cp = 0; cp < 4; ++cp) {
      int kbase = (cp * Cc + ci) * 3;
      h16 hv = (h16)vals[cp];
      sT[ti * T_LD + kbase + 2] = hv;
      if (ti + 1 < Tt) sT[(ti + 1) * T_LD + kbase + 1] = hv;
      if (ti + 2 < Tt) sT[(ti + 2) * T_LD + kbase + 0] = hv;
    }
    sOut[ci * AT_LD + ti] = raw;  // residual = input_f
  }
  wait_async0();
  __syncthreads();

  // q = qW @ im2col : [32 x 64], K=768, one 16x16 tile per wave
  {
    int mt = wave >> 2, nt = wave & 3;
    v8f accq = {}, acck = {};
    for (int kt = 0; kt < KK / 32; ++kt) {
      v16h bt = fragA(sT, T_LD, nt * 16, kt * 32, lane);  // B[k][n]=sT[n][k]
      v16h aq = fragA(sWa, W_LD, mt * 16, kt * 32, lane);
      accq = wmma16(aq, bt, accq);
      v16h ak = fragA(sWb, W_LD, mt * 16, kt * 32, lane);
      acck = wmma16(ak, bt, acck);
    }
    storeD_hT(sQt, QT_LD, mt * 16, nt * 16, lane, accq);  // q^T [t][h]
    storeD_hT(sKt, QT_LD, mt * 16, nt * 16, lane, acck);  // k^T [t][h]
  }
  __syncthreads();

  // overlap: DMA vW768 -> sWa, packed oW/ffW1/ffW2 -> sWb during attention
  for (int e = tid; e < 3104; e += 256)
    async_ld16(&sWa[e * 8], wh + OFF_VW + e * 8);
  for (int e = tid; e < PK_HALVES / 8; e += 256)
    async_ld16(&sWb[e * 8], wh + OFF_PK + e * 8);

  // attn logits [64 x 64] = q k^T, K=32 (two tiles per wave)
  for (int p = 0; p < 2; ++p) {
    int tt = wave + p * 8;
    int mt = tt >> 2, nt = tt & 3;
    v8f acc = {};
    v16h a = fragA(sQt, QT_LD, mt * 16, 0, lane);  // A[i][h]
    v16h b = fragA(sKt, QT_LD, nt * 16, 0, lane);  // B[h][j]=k^T[j][h]
    acc = wmma16(a, b, acc);
    storeD_f(sAttn, AT_LD, mt * 16, nt * 16, lane, acc);
  }
  __syncthreads();

  // causal softmax (exact reference semantics: logits - 1000*upper)
  if (tid < Tt) {
    int i = tid;
    float mx = -1e30f;
    for (int j = 0; j < Tt; ++j) {
      float v = sAttn[i * AT_LD + j] - (j > i ? 1000.f : 0.f);
      sAttn[i * AT_LD + j] = v;
      mx = fmaxf(mx, v);
    }
    float sum = 0.f;
    for (int j = 0; j < Tt; ++j) {
      float e = __expf(sAttn[i * AT_LD + j] - mx);
      sAttn[i * AT_LD + j] = e;
      sum += e;
    }
    float inv = 1.f / sum;
    for (int j = 0; j < Tt; ++j)
      sAH[i * AH_LD + j] = (h16)(sAttn[i * AT_LD + j] * inv);
  }
  wait_async0();
  __syncthreads();

  // v = vW768 @ im2col : [32 x 64], K=768
  {
    int mt = wave >> 2, nt = wave & 3;
    v8f acc = {};
    for (int kt = 0; kt < KK / 32; ++kt) {
      v16h a = fragA(sWa, W_LD, mt * 16, kt * 32, lane);
      v16h b = fragA(sT, T_LD, nt * 16, kt * 32, lane);
      acc = wmma16(a, b, acc);
    }
    storeD_h(sV, AH_LD, mt * 16, nt * 16, lane, acc);  // [h][t]
  }
  __syncthreads();

  // attn_out [32 x 64] = v @ P^T, K=64 ; probs row-major => B[k=j][n=i]=P[i][j]
  {
    int mt = wave >> 2, nt = wave & 3;
    v8f acc = {};
    for (int kt = 0; kt < 2; ++kt) {
      v16h a = fragA(sV, AH_LD, mt * 16, kt * 32, lane);
      v16h b = fragA(sAH, AH_LD, nt * 16, kt * 32, lane);
      acc = wmma16(a, b, acc);
    }
    storeD_hT(sQt, QT_LD, mt * 16, nt * 16, lane, acc);  // attn_out^T [t][h]
  }
  __syncthreads();

  // out_f = input_f + oW @ attn_out : [64 x 64], K=32
  for (int p = 0; p < 2; ++p) {
    int tt = wave + p * 8;
    int mt = tt >> 2, nt = tt & 3;
    v8f acc = loadC_f(sOut, AT_LD, mt * 16, nt * 16, lane);
    v16h a = fragA(sWb, 40, mt * 16, 0, lane);           // oW [c][h]
    v16h b = fragA(sQt, QT_LD, nt * 16, 0, lane);        // B[h][t]=AO^T[t][h]
    acc = wmma16(a, b, acc);
    storeD_f(sOut, AT_LD, mt * 16, nt * 16, lane, acc);
  }
  __syncthreads();
  // out_f -> f16 transposed [t][c] (reuse sAH)
  for (int e = tid; e < Cc * Tt; e += 256) {
    int ci = e >> 6, ti = e & 63;
    sAH[ti * AH_LD + ci] = (h16)sOut[ci * AT_LD + ti];
  }
  __syncthreads();

  // h = relu(ffW1 @ out_f + b1) : [64 x 64], K=64 (store transposed)
  for (int p = 0; p < 2; ++p) {
    int tt = wave + p * 8;
    int mt = tt >> 2, nt = tt & 3;
    v8f acc = {};
    for (int kt = 0; kt < 2; ++kt) {
      v16h a = fragA(sWb + 2560, 72, mt * 16, kt * 32, lane);
      v16h b = fragA(sAH, AH_LD, nt * 16, kt * 32, lane);  // B[c][t]=outT[t][c]
      acc = wmma16(a, b, acc);
    }
    int nn = (lane & 15) + nt * 16;
    int mo = (lane & 16) ? 8 : 0;
#pragma unroll
    for (int r = 0; r < 8; ++r) {
      int m = mt * 16 + mo + r;
      sHt[nn * AH_LD + m] = (h16)fmaxf(acc[r] + ffb1[m], 0.f);
    }
  }
  __syncthreads();

  // out_f += ffW2 @ h + b2 : [64 x 64], K=64
  for (int p = 0; p < 2; ++p) {
    int tt = wave + p * 8;
    int mt = tt >> 2, nt = tt & 3;
    v8f acc = loadC_f(sOut, AT_LD, mt * 16, nt * 16, lane);
    for (int kt = 0; kt < 2; ++kt) {
      v16h a = fragA(sWb + 7168, 72, mt * 16, kt * 32, lane);
      v16h b = fragA(sHt, AH_LD, nt * 16, kt * 32, lane);  // B[d][t]=h^T[t][d]
      acc = wmma16(a, b, acc);
    }
    int nn = (lane & 15) + nt * 16;
    int mo = (lane & 16) ? 8 : 0;
#pragma unroll
    for (int r = 0; r < 8; ++r) {
      int m = mt * 16 + mo + r;
      sOut[m * AT_LD + nn] = acc[r] + ffb2[m];
    }
  }
  __syncthreads();

  // write out[bi][ci][ni][ti] = out_f[ci][ti]  (coalesced in ti)
  for (int e = tid; e < Cc * Tt; e += 256) {
    int ci = e >> 6, ti = e & 63;
    out[(((size_t)bi * Cc + ci) * Nn + ni) * Tt + ti] = sOut[ci * AT_LD + ti];
  }
}

extern "C" void kernel_launch(void* const* d_in, const int* in_sizes, int n_in,
                              void* d_out, int out_size, void* d_ws,
                              size_t ws_size, hipStream_t stream) {
  const float* x = (const float*)d_in[0];
  const float* tn_g = (const float*)d_in[1];
  const float* tn_b = (const float*)d_in[2];
  const float* sn_g = (const float*)d_in[3];
  const float* sn_b = (const float*)d_in[4];
  const float* stn_g = (const float*)d_in[5];
  const float* stn_b = (const float*)d_in[6];
  const float* qW = (const float*)d_in[7];
  const float* kW = (const float*)d_in[8];
  const float* vW = (const float*)d_in[9];
  const float* oW = (const float*)d_in[10];
  const float* ffW1 = (const float*)d_in[11];
  const float* ffb1 = (const float*)d_in[12];
  const float* ffW2 = (const float*)d_in[13];
  const float* ffb2 = (const float*)d_in[14];

  float* ws = (float*)d_ws;
  float* tn_mu = ws;                     // 12800
  float* tn_rs = tn_mu + Cc * Nn;        // 12800
  float* sn_mu = tn_rs + Cc * Nn;        // 65536
  float* sn_rs = sn_mu + Bq * Cc * Tt;   // 65536
  float* stn_mu = sn_rs + Bq * Cc * Tt;  // 1024
  float* stn_rs = stn_mu + Bq * Cc;      // 1024  (ends at 158720 floats)
  h16* wh = (h16*)(ws + 160000);         // f16 weight blob (86272 halves)

  cvt_w<<<97, 256, 0, stream>>>(qW, kW, vW, oW, ffW1, ffW2, wh);
  stats_tn<<<Cc * Nn, 256, 0, stream>>>(x, tn_mu, tn_rs);
  stats_sn<<<(Bq * Cc * Tt) / 256, 256, 0, stream>>>(x, sn_mu, sn_rs);
  stats_stn<<<Bq * Cc, 256, 0, stream>>>(x, stn_mu, stn_rs);
  attn_seq<<<Bq * Nn, 256, 0, stream>>>(
      x, tn_g, tn_b, sn_g, sn_b, stn_g, stn_b, ffb1, ffb2, tn_mu, tn_rs, sn_mu,
      sn_rs, stn_mu, stn_rs, wh, (float*)d_out);
}
